// SemCLIPVisionEmbeddings_75316546502747
// MI455X (gfx1250) — matrix operand order repaired and checked by reading, not compiled, not run on hardware
//
#include <hip/hip_runtime.h>
#include <hip/hip_bf16.h>

// ---------- types ----------
typedef __attribute__((ext_vector_type(16))) __bf16 v16bf;
typedef __attribute__((ext_vector_type(8)))  float  v8f;
typedef __attribute__((ext_vector_type(4)))  int    v4i;

typedef __attribute__((address_space(1))) v4i av4i_g;   // global int4
typedef __attribute__((address_space(3))) v4i av4i_l;   // LDS int4

union frag16 { uint4 q[2]; v16bf v; };

__device__ __forceinline__ unsigned short f2bf(float f) {
  union { float f; unsigned u; } a; a.f = f;
  unsigned u = a.u;
  unsigned r = u + 0x7FFFu + ((u >> 16) & 1u);   // round to nearest even
  return (unsigned short)(r >> 16);
}

// ---------- gfx1250 async global->LDS copy (ASYNCcnt path) ----------
__device__ __forceinline__ void async_cp_b128(const void* gptr, unsigned lds_byte_off) {
#if __has_builtin(__builtin_amdgcn_global_load_async_to_lds_b128)
  __builtin_amdgcn_global_load_async_to_lds_b128(
      (av4i_g*)gptr,
      (av4i_l*)(unsigned long)lds_byte_off,
      0, 0);
#else
  asm volatile("global_load_async_to_lds_b128 %0, %1, off"
               :: "v"(lds_byte_off), "v"(gptr) : "memory");
#endif
}

__device__ __forceinline__ void wait_async0() {
#if __has_builtin(__builtin_amdgcn_s_wait_asynccnt)
  __builtin_amdgcn_s_wait_asynccnt(0);
#else
  asm volatile("s_wait_asynccnt 0x0" ::: "memory");
#endif
}

// ---------- kernel 1: row LayerNorm -> bf16 ----------
__global__ __launch_bounds__(256) void ln_bf16(const float* __restrict__ x,
                                               const float* __restrict__ w,
                                               const float* __restrict__ b,
                                               unsigned short* __restrict__ xb)
{
  const int t = threadIdx.x;
  const long row = blockIdx.x;
  const float* xr = x + row * 3072;

  float4 v[3];
  float s = 0.f, s2 = 0.f;
#pragma unroll
  for (int i = 0; i < 3; ++i) {
    v[i] = *(const float4*)(xr + (i * 256 + t) * 4);
    s  += v[i].x + v[i].y + v[i].z + v[i].w;
    s2 += v[i].x * v[i].x + v[i].y * v[i].y + v[i].z * v[i].z + v[i].w * v[i].w;
  }

  __shared__ float ssum[256], ssq[256];
  ssum[t] = s; ssq[t] = s2;
  __syncthreads();
  for (int off = 128; off > 0; off >>= 1) {
    if (t < off) { ssum[t] += ssum[t + off]; ssq[t] += ssq[t + off]; }
    __syncthreads();
  }
  const float mean = ssum[0] * (1.0f / 3072.0f);
  const float var  = ssq[0]  * (1.0f / 3072.0f) - mean * mean;
  const float rstd = rsqrtf(var + 1e-5f);

  unsigned short* orow = xb + row * 3072;
#pragma unroll
  for (int i = 0; i < 3; ++i) {
    const int base = (i * 256 + t) * 4;
    float4 wv = *(const float4*)(w + base);
    float4 bv = *(const float4*)(b + base);
    ushort4 o;
    o.x = f2bf((v[i].x - mean) * rstd * wv.x + bv.x);
    o.y = f2bf((v[i].y - mean) * rstd * wv.y + bv.y);
    o.z = f2bf((v[i].z - mean) * rstd * wv.z + bv.z);
    o.w = f2bf((v[i].w - mean) * rstd * wv.w + bv.w);
    *(ushort4*)(orow + base) = o;
  }
}

// ---------- kernel 2: W_patch fp32 -> bf16 ----------
__global__ __launch_bounds__(256) void wcvt(const float* __restrict__ W,
                                            unsigned short* __restrict__ wb)
{
  const long i = ((long)blockIdx.x * 256 + threadIdx.x) * 4;
  float4 v = *(const float4*)(W + i);
  ushort4 o;
  o.x = f2bf(v.x); o.y = f2bf(v.y); o.z = f2bf(v.z); o.w = f2bf(v.w);
  *(ushort4*)(wb + i) = o;
}

// ---------- kernel 3: bf16 WMMA GEMM with async-to-LDS double buffering ----------
// out[1+m][n] = sum_k xb[m][k]*wb[n][k] + (bbox[m] . W_pos[n]) + b_pos[n]
// block tile 128x128, K-step 32, 8 waves (4x2), wave tile 32x64
#define KSTEPS 96            // 3072 / 32
#define LDSW   40            // padded row stride in bf16 elems (80B)
#define TILEB  (128 * LDSW * 2)       // 10240 bytes per tile buffer
#define TILEE  (128 * LDSW)           // 5120 ushort elems per tile buffer

__global__ __launch_bounds__(256) void gemm_wmma(const unsigned short* __restrict__ xb,
                                                 const unsigned short* __restrict__ wb,
                                                 const float* __restrict__ bbox,
                                                 const float* __restrict__ Wp,
                                                 const float* __restrict__ bp,
                                                 float* __restrict__ out)
{
  extern __shared__ unsigned short smem[];   // dynamic LDS only: base offset 0
  // layout (bytes): A0 @ 0, A1 @ 10240, B0 @ 20480, B1 @ 30720

  const int t = threadIdx.x;
  const int nBase = blockIdx.x * 128;   // N (D=1024) tiles vary fastest -> W stays in L2
  const int mBase = blockIdx.y * 128;

  // global->LDS copy assignment: 512 chunks of 16B per tile, 2 per thread
  const int row0 = t >> 2;              // 0..63
  const int colE = (t & 3) * 8;         // bf16 elem offset 0/8/16/24
  const int row1 = row0 + 64;
  const unsigned ls0b = (unsigned)(row0 * LDSW + colE) * 2;   // LDS byte offsets
  const unsigned ls1b = (unsigned)(row1 * LDSW + colE) * 2;

  const unsigned short* pa0 = xb + (size_t)(mBase + row0) * 3072 + colE;
  const unsigned short* pa1 = xb + (size_t)(mBase + row1) * 3072 + colE;
  const unsigned short* pb0 = wb + (size_t)(nBase + row0) * 3072 + colE;
  const unsigned short* pb1 = wb + (size_t)(nBase + row1) * 3072 + colE;

  // wave layout
  const int w    = t >> 5;
  const int lane = t & 31;
  const int wm = w & 3, wn = w >> 2;
  const int rowW = wm * 32;             // wave rows within block tile
  const int colW = wn * 64;             // wave cols within block tile
  const int lrow = lane & 15;
  const int hi   = lane >> 4;           // 0: low half lanes, 1: high half lanes

  v8f acc[2][4];
#pragma unroll
  for (int i = 0; i < 2; ++i)
#pragma unroll
    for (int j = 0; j < 4; ++j)
#pragma unroll
      for (int r = 0; r < 8; ++r) acc[i][j][r] = 0.0f;

  // ---- preload stage 0 (buffer 0) ----
  async_cp_b128(pa0, 0u       + ls0b);
  async_cp_b128(pa1, 0u       + ls1b);
  async_cp_b128(pb0, 2u*TILEB + ls0b);
  async_cp_b128(pb1, 2u*TILEB + ls1b);
  pa0 += 32; pa1 += 32; pb0 += 32; pb1 += 32;
  wait_async0();
  __syncthreads();

  for (int kk = 0; kk < KSTEPS; ++kk) {
    const int buf = kk & 1;

    if (kk + 1 < KSTEPS) {              // async-fill the other buffer
      const unsigned nb = (unsigned)(buf ^ 1);
      async_cp_b128(pa0, nb * TILEB            + ls0b);
      async_cp_b128(pa1, nb * TILEB            + ls1b);
      async_cp_b128(pb0, (2u + nb) * TILEB     + ls0b);
      async_cp_b128(pb1, (2u + nb) * TILEB     + ls1b);
      pa0 += 32; pa1 += 32; pb0 += 32; pb1 += 32;
    }
    if (kk + 2 < KSTEPS) {              // prefetch stage kk+2 into caches
      __builtin_prefetch(pa0, 0, 1);
      __builtin_prefetch(pb0, 0, 1);
    }

    // LDS -> WMMA fragments (per CDNA5 16-bit A / B lane layouts)
    frag16 af[2], bf[4];
    const unsigned short* As = smem + (unsigned)buf * TILEE;
    const unsigned short* Bs = smem + (2u + (unsigned)buf) * TILEE;
#pragma unroll
    for (int mi = 0; mi < 2; ++mi) {
      const unsigned short* p = As + (rowW + mi * 16 + lrow) * LDSW + hi * 8;
      af[mi].q[0] = *(const uint4*)p;          // K 0..7   (or 8..15)
      af[mi].q[1] = *(const uint4*)(p + 16);   // K 16..23 (or 24..31)
    }
#pragma unroll
    for (int ni = 0; ni < 4; ++ni) {
      const unsigned short* p = Bs + (colW + ni * 16 + lrow) * LDSW + hi * 8;
      bf[ni].q[0] = *(const uint4*)p;          // K 0..7   (or 8..15)
      bf[ni].q[1] = *(const uint4*)(p + 16);   // K 16..23 (or 24..31)
    }

#pragma unroll
    for (int mi = 0; mi < 2; ++mi)
#pragma unroll
      for (int ni = 0; ni < 4; ++ni)
        acc[mi][ni] = __builtin_amdgcn_wmma_f32_16x16x32_bf16(
            false, af[mi].v, false, bf[ni].v, (short)0, acc[mi][ni], false, false);

    if (kk + 1 < KSTEPS) wait_async0();  // my fills landed; barrier covers the rest
    __syncthreads();
  }

  // ---- fused epilogue: C/D layout + rank-4 positional projection + bias ----
  float4 wp[4]; float bpv[4];
#pragma unroll
  for (int ni = 0; ni < 4; ++ni) {
    const int gn = nBase + colW + ni * 16 + lrow;
    wp[ni]  = *(const float4*)(Wp + (size_t)gn * 4);
    bpv[ni] = bp[gn];
  }
#pragma unroll
  for (int mi = 0; mi < 2; ++mi) {
    const int gmBase = mBase + rowW + mi * 16 + hi * 8;
    float4 bbv[8];
#pragma unroll
    for (int r = 0; r < 8; ++r)
      bbv[r] = *(const float4*)(bbox + (size_t)(gmBase + r) * 4);
#pragma unroll
    for (int ni = 0; ni < 4; ++ni) {
      const int gn = nBase + colW + ni * 16 + lrow;
      float* op = out + (size_t)(1 + gmBase) * 1024 + gn;
#pragma unroll
      for (int r = 0; r < 8; ++r) {
        const float pos = bpv[ni] + bbv[r].x * wp[ni].x + bbv[r].y * wp[ni].y
                        + bbv[r].z * wp[ni].z + bbv[r].w * wp[ni].w;
        op[(size_t)r * 1024] = acc[mi][ni][r] + pos;
      }
    }
  }
}

// ---------- kernel 4: class row ----------
__global__ __launch_bounds__(256) void class_row(const float* __restrict__ cls,
                                                 const float* __restrict__ cpos,
                                                 float* __restrict__ out)
{
  const int d = blockIdx.x * 256 + threadIdx.x;
  if (d < 1024) out[d] = cls[d] + cpos[d];
}

// ---------- launcher ----------
extern "C" void kernel_launch(void* const* d_in, const int* in_sizes, int n_in,
                              void* d_out, int out_size, void* d_ws, size_t ws_size,
                              hipStream_t stream) {
  (void)in_sizes; (void)n_in; (void)out_size; (void)ws_size;
  const float* patches   = (const float*)d_in[0];   // [16384,3072]
  const float* bbox      = (const float*)d_in[1];   // [16384,4]
  const float* ln_w      = (const float*)d_in[2];   // [3072]
  const float* ln_b      = (const float*)d_in[3];   // [3072]
  const float* W_patch   = (const float*)d_in[4];   // [1024,3072]
  const float* class_emb = (const float*)d_in[5];   // [1024]
  const float* W_pos     = (const float*)d_in[6];   // [1024,4]
  const float* b_pos     = (const float*)d_in[7];   // [1024]
  const float* class_pos = (const float*)d_in[8];   // [1024]
  float* out = (float*)d_out;                       // [16385,1024]

  unsigned short* xb = (unsigned short*)d_ws;                   // 16384*3072 bf16
  unsigned short* wb = xb + (size_t)16384 * 3072;               // 1024*3072 bf16

  ln_bf16<<<16384, 256, 0, stream>>>(patches, ln_w, ln_b, xb);
  wcvt<<<(1024 * 3072) / (256 * 4), 256, 0, stream>>>(W_patch, wb);

  dim3 g(8, 128);                                   // x = N tiles (1024/128), y = M tiles
  gemm_wmma<<<g, 256, 4 * TILEB, stream>>>(xb, wb, bbox, W_pos, b_pos, out);

  class_row<<<4, 256, 0, stream>>>(class_emb, class_pos, out);
}
